// Upsample_30185030157082
// MI455X (gfx1250) — matrix operand order repaired and checked by reading, not compiled, not run on hardware
//
#include <hip/hip_runtime.h>

// MI455X / gfx1250, wave32. Upsample2x + ternary 3x3 conv as 4 parity-folded
// implicit GEMMs: M=co=256, K=ci*4=1024, N=16*64*64 per parity.
// f16 WMMA 16x16x32, f32 accumulate. 64x32 wave tiles (1.5 frag-loads/WMMA),
// padded LDS row cache, one-step register pipeline for global loads.

typedef __attribute__((ext_vector_type(16))) _Float16 v16h;
typedef __attribute__((ext_vector_type(8)))  float    v8f;
typedef __attribute__((ext_vector_type(4)))  float    v4f;
typedef __attribute__((ext_vector_type(4)))  _Float16 v4h;

#define B_  16
#define C_  256
#define KTOT 1024                    // ci(256) * {dy,dx}(4)

#define XH_ELEMS (B_*C_*64*64)       // 16,777,216 halfs (33.5 MB)
#define WF_ELEMS (4*C_*KTOT)         // 1,048,576 halfs (2 MB)
#define WF_BYTE_OFFSET ((size_t)XH_ELEMS * 2)

// ---------------- kernel 1: x (f32) -> xh (f16) ----------------
__global__ void cast_x_f16(const float* __restrict__ x, _Float16* __restrict__ xh) {
  int i = (blockIdx.x * blockDim.x + threadIdx.x) * 4;
  v4f v = *(const v4f*)(x + i);
  v4h h;
  h[0] = (_Float16)v[0]; h[1] = (_Float16)v[1];
  h[2] = (_Float16)v[2]; h[3] = (_Float16)v[3];
  *(v4h*)(xh + i) = h;
}

// ------------- kernel 2: ternary quantize + parity fold -------------
__device__ __forceinline__ float tern(float w) {
  return w > 0.001f ? 1.0f : (w < -0.001f ? -1.0f : 0.0f);
}

// Wf[p][co][k], k = ci*4 + dy*2 + dx; flat index == thread id
__global__ void fold_weights(const float* __restrict__ w, _Float16* __restrict__ wf) {
  int t  = blockIdx.x * 256 + threadIdx.x;       // 2^20 threads
  int p  = t >> 18;
  int co = (t >> 10) & (C_ - 1);
  int k  = t & (KTOT - 1);
  int ci = k >> 2;
  int dy = (k >> 1) & 1;
  int dx = k & 1;
  int ph = p >> 1, pw = p & 1;
  int kh0 = (ph == 0) ? (dy ? 1 : 0) : (dy ? 2 : 0);
  int kh1 = (ph == 0) ? (dy ? 2 : 0) : (dy ? 2 : 1);
  int kw0 = (pw == 0) ? (dx ? 1 : 0) : (dx ? 2 : 0);
  int kw1 = (pw == 0) ? (dx ? 2 : 0) : (dx ? 2 : 1);
  float s = 0.0f;
  for (int kh = kh0; kh <= kh1; ++kh)
    for (int kw = kw0; kw <= kw1; ++kw)
      s += tern(w[((co * C_ + ci) * 3 + kh) * 3 + kw]);
  wf[t] = (_Float16)s;   // folded sums in [-4,4]: exact in f16
}

// ---------------- kernel 3: WMMA implicit GEMM ----------------
// grid = (1024 N-tiles, 4 parities), block = 256 (8 waves as 4M x 2N).
// WG tile: M=256 (all co), N=64.  Wave tile: 64x32 -> 8 WMMA / 12 frag loads.
#define XROW  72   // padded row: halo at idx 3 (col -1) / idx 68 (col 64); data 4..67
#define XROWS 17   // 16 real rows + 1 trash row for out-of-range stores

__global__ __launch_bounds__(256)
void upconv_gemm(const _Float16* __restrict__ xh,
                 const _Float16* __restrict__ wf,
                 const float*    __restrict__ bias,
                 float*          __restrict__ out) {
  __shared__ __align__(16) _Float16 sA[256 * 32];            // [co][k]   16 KB
  __shared__ __align__(16) _Float16 sB[64 * 32];             // [n][k]     4 KB
  __shared__ __align__(16) _Float16 sXall[2 * XROWS * XROW]; // ping-pong row cache

  const int p  = blockIdx.y;
  const int ph = p >> 1, pw = p & 1;
  const int nt = blockIdx.x;
  const int b  = nt >> 6;          // batch
  const int m  = nt & 63;          // oh = 2m + ph

  const int t    = threadIdx.x;
  const int lane = t & 31;
  const int wave = t >> 5;
  const int wm   = wave >> 1;      // 0..3: wave M block = wm*64
  const int wn   = wave & 1;       // 0..1: wave N block = wn*32
  const int l15  = lane & 15;
  const int lHi  = lane >> 4;

  v8f acc[4][2] = {};

  // A staging: 1 thread per co row (32 halfs = 64B)
  const _Float16* wfRow = wf + (size_t)(p * C_ + t) * KTOT;
  _Float16* sArow = sA + t * 32;

  // Phase-0 (row cache): 16 rows (8 ci x 2 dy), 16 threads x 4 halfs per row
  const int xr   = t >> 4;                        // 0..15
  const int xciL = xr >> 1;
  const int xdy  = xr & 1;
  const int xseg = t & 15;
  const int xrow = m + xdy - 1 + ph;              // -1..64
  const bool rowOk = (unsigned)xrow < 64u;
  const int xrStore = rowOk ? xr : 16;            // invalid rows -> trash row
  const int xrowC   = rowOk ? xrow : 0;           // clamp: load stays in-bounds
  const size_t xgBase =
      ((size_t)(b * C_ + xciL) * 64 + xrowC) * 64 + xseg * 4;

  // Phase-1 (B build) params
  const int bn  = t & 63;
  const int grp = t >> 6;

  // Pre-zero row cache once: halo slots + permanently-invalid rows stay 0.
  for (int i = t; i < (2 * XROWS * XROW) / 4; i += 256)
    ((v4h*)sXall)[i] = (v4h)0;
  __syncthreads();

  // ---- preload K-step 0 into registers ----
  v4f wA0 = *(const v4f*)(wfRow);
  v4f wA1 = *(const v4f*)(wfRow + 8);
  v4f wA2 = *(const v4f*)(wfRow + 16);
  v4f wA3 = *(const v4f*)(wfRow + 24);
  v4h xv  = *(const v4h*)(xh + xgBase);

#pragma unroll 4
  for (int kb = 0; kb < 32; ++kb) {
    _Float16* sXc = sXall + (kb & 1) * (XROWS * XROW);

    // phase 0: commit current activation rows (other sX buffer still live)
    *(v4h*)(sXc + xrStore * XROW + 4 + xseg * 4) = xv;

    __syncthreads();   // sX ready; prior frag reads of sA/sB done

    // phase 1: sA from regs; gather sB[n][k] from padded row cache
    *(v4f*)(sArow)      = wA0;
    *(v4f*)(sArow + 8)  = wA1;
    *(v4f*)(sArow + 16) = wA2;
    *(v4f*)(sArow + 24) = wA3;
    union { _Float16 h[8]; v4f f; } pk;
#pragma unroll
    for (int j = 0; j < 8; ++j) {
      int kk  = grp * 8 + j;
      int ciL = kk >> 2;
      int dy  = (kk >> 1) & 1;
      int dx  = kk & 1;
      pk.h[j] = sXc[(ciL * 2 + dy) * XROW + bn + dx + pw + 3];
    }
    *(v4f*)(sB + bn * 32 + grp * 8) = pk.f;

    // issue NEXT K-step's global loads; wait lands after barrier + 8 WMMAs
    const int kn = (kb + 1) & 31;
    wA0 = *(const v4f*)(wfRow + kn * 32);
    wA1 = *(const v4f*)(wfRow + kn * 32 + 8);
    wA2 = *(const v4f*)(wfRow + kn * 32 + 16);
    wA3 = *(const v4f*)(wfRow + kn * 32 + 24);
    xv  = *(const v4h*)(xh + xgBase + (size_t)(kn * 8) * 4096);
    __builtin_prefetch(wfRow + ((kb + 2) & 31) * 32, 0, 3);

    __syncthreads();   // sA/sB ready

    // ---- fragment loads per documented wave32 layouts ----
    union Frag { v16h h; v4f f[2]; };
    Frag b0, b1;
    const _Float16* bB0 = sB + (wn * 32 + l15) * 32 + lHi * 8;
    const _Float16* bB1 = sB + (wn * 32 + 16 + l15) * 32 + lHi * 8;
    b0.f[0] = *(const v4f*)(bB0);
    b0.f[1] = *(const v4f*)(bB0 + 16);
    b1.f[0] = *(const v4f*)(bB1);
    b1.f[1] = *(const v4f*)(bB1 + 16);

#pragma unroll
    for (int i = 0; i < 4; ++i) {
      Frag ai;
      const _Float16* aB = sA + (wm * 64 + i * 16 + l15) * 32;
      ai.f[0] = *(const v4f*)(aB + lHi * 8);
      ai.f[1] = *(const v4f*)(aB + 16 + lHi * 8);
      acc[i][0] = __builtin_amdgcn_wmma_f32_16x16x32_f16(false, ai.h, false, b0.h, (short)0, acc[i][0], false, false);
      acc[i][1] = __builtin_amdgcn_wmma_f32_16x16x32_f16(false, ai.h, false, b1.h, (short)0, acc[i][1], false, false);
    }
  }

  // ---- epilogue: D layout lanes0-15 M=r, lanes16-31 M=r+8, N=lane%16 ----
  const int oh = 2 * m + ph;
  const size_t outB = (size_t)b * C_ * 128 * 128;
#pragma unroll
  for (int i = 0; i < 4; ++i) {
    int coBase = wm * 64 + i * 16 + lHi * 8;
#pragma unroll
    for (int j = 0; j < 2; ++j) {
      int nL = wn * 32 + j * 16 + l15;
      int ow = 2 * nL + pw;
      v8f a = acc[i][j];
#pragma unroll
      for (int r = 0; r < 8; ++r) {
        int co = coBase + r;
        out[outB + ((size_t)co * 128 + oh) * 128 + ow] = a[r] + bias[co];
      }
    }
  }
}

extern "C" void kernel_launch(void* const* d_in, const int* in_sizes, int n_in,
                              void* d_out, int out_size, void* d_ws, size_t ws_size,
                              hipStream_t stream) {
  (void)in_sizes; (void)n_in; (void)out_size; (void)ws_size;
  const float* x    = (const float*)d_in[0];   // [16,256,64,64]
  const float* w    = (const float*)d_in[3];   // [256,256,3,3]
  const float* bias = (const float*)d_in[4];   // [256]
  float* out = (float*)d_out;                  // [16,256,128,128]

  _Float16* xh = (_Float16*)d_ws;
  _Float16* wf = (_Float16*)((char*)d_ws + WF_BYTE_OFFSET);

  cast_x_f16 <<<XH_ELEMS / (4 * 256), 256, 0, stream>>>(x, xh);
  fold_weights<<<WF_ELEMS / 256,       256, 0, stream>>>(w, wf);

  dim3 grid(1024, 4);   // N tiles x parity
  upconv_gemm<<<grid, 256, 0, stream>>>(xh, wf, bias, out);
}
